// AttentiveFPEncoder_27565100106036
// MI455X (gfx1250) — matrix operand order repaired and checked by reading, not compiled, not run on hardware
//
#include <hip/hip_runtime.h>
#include <math.h>

// ---------------------------------------------------------------------------
// AttentiveFP encoder for MI455X (gfx1250, wave32).
//  - dense GEMMs: v_wmma_f32_16x16x32_bf16, 16x64 tile per wave
//  - GRU cells fully fused in-register (6 WMMA accumulators per wave),
//    weights staged to LDS with global_load_async_to_lds_b128 (ASYNCcnt)
//  - GATEConv edge attention fused (never materializes [E,256])
// ---------------------------------------------------------------------------

typedef __bf16 bf16_t;
typedef __attribute__((ext_vector_type(16))) __bf16 v16bf;
typedef __attribute__((ext_vector_type(8)))  __bf16 v8bf;
typedef __attribute__((ext_vector_type(8)))  float  v8f;

#define HD   256      // hidden dim
#define GN   2048     // graphs
#define SLOPE 0.01f

// order-preserving float <-> uint for atomicMax-based segment max
static __device__ __forceinline__ unsigned encf(float f) {
  unsigned u = __float_as_uint(f);
  return (u & 0x80000000u) ? ~u : (u | 0x80000000u);
}
static __device__ __forceinline__ float decf(unsigned u) {
  return __uint_as_float((u & 0x80000000u) ? (u ^ 0x80000000u) : ~u);
}
#define ENC_NEG_INF 0x007FFFFFu   // encf(-inf)

// ---------------------------------------------------------------------------
// fp32 -> bf16 convert with column padding (pad columns zero-filled)
// ---------------------------------------------------------------------------
__global__ void cvt_pad_bf16(const float* __restrict__ s, bf16_t* __restrict__ d,
                             long long R, int C, int Cp) {
  long long t = (long long)blockIdx.x * blockDim.x + threadIdx.x;
  if (t >= R * (long long)Cp) return;
  long long r = t / Cp; int c = (int)(t % Cp);
  d[t] = (c < C) ? (bf16_t)s[r * (long long)C + c] : (bf16_t)0.0f;
}

// load one 16x32 bf16 A fragment (row-major, lda=K)
static __device__ __forceinline__ v16bf load_afrag(const bf16_t* pa, int k0) {
  v8bf lo = *(const v8bf*)(pa + k0);
  v8bf hi = *(const v8bf*)(pa + k0 + 8);
  v16bf a;
#pragma unroll
  for (int i = 0; i < 8; ++i) { a[i] = lo[i]; a[8 + i] = hi[i]; }
  return a;
}

// ---------------------------------------------------------------------------
// Generic bf16 WMMA GEMM: C[M,Nc] = act(A[M,K] @ W[Nc,K]^T + bias)
// One wave computes a 16x64 tile (4 accumulators, A fragment reused 4x).
// K multiple of 32, Nc multiple of 64. act: 0 none, 1 lrelu, 2 relu.
// ---------------------------------------------------------------------------
__global__ __launch_bounds__(256)
void gemm4_bf16_wmma(const bf16_t* __restrict__ A, const bf16_t* __restrict__ W,
                     const float* __restrict__ bias, float* __restrict__ C,
                     bf16_t* __restrict__ Cbf, int M, int Nc, int K, int act) {
  const int wave = threadIdx.x >> 5;
  const int lane = threadIdx.x & 31;
  const int hf   = lane >> 4;
  const int l15  = lane & 15;
  const int n0 = blockIdx.x * 64;
  const int m0 = (blockIdx.y * 8 + wave) * 16;
  if (m0 >= M) return;

  int rowA = m0 + l15; if (rowA >= M) rowA = M - 1;     // clamp loads, guard stores
  const bf16_t* pa = A + (size_t)rowA * K + hf * 16;    // A: 16 contiguous K / lane
  const bf16_t* pb0 = W + (size_t)(n0 + l15) * K + hf * 16;  // B: 16 contiguous K / lane
  const size_t bstep = (size_t)16 * K;

  v8f c[4];
#pragma unroll
  for (int q = 0; q < 4; ++q)
#pragma unroll
    for (int r = 0; r < 8; ++r) c[q][r] = 0.f;

  for (int k0 = 0; k0 < K; k0 += 32) {
    v16bf a = load_afrag(pa, k0);
#pragma unroll
    for (int q = 0; q < 4; ++q) {
      v16bf b = *(const v16bf*)(pb0 + q * bstep + k0);
      c[q] = __builtin_amdgcn_wmma_f32_16x16x32_bf16(false, a, false, b,
                                                     (short)0, c[q], false, false);
    }
  }
#pragma unroll
  for (int q = 0; q < 4; ++q) {
    const int col = n0 + q * 16 + l15;
    const float bv = bias ? bias[col] : 0.0f;
#pragma unroll
    for (int r = 0; r < 8; ++r) {
      int row = m0 + hf * 8 + r;
      if (row < M) {
        float v = c[q][r] + bv;
        if (act == 1)      v = v >= 0.f ? v : SLOPE * v;
        else if (act == 2) v = v >  0.f ? v : 0.f;
        C[(size_t)row * Nc + col] = v;
        if (Cbf) Cbf[(size_t)row * Nc + col] = (bf16_t)v;
      }
    }
  }
}

// ---------------------------------------------------------------------------
// Fully fused GRU cell + relu:
//   xout = relu( (1-z)*n + z*hprev ),  gates from
//   gi = hin @ wih^T + bih   (hin = elu(h) bf16)
//   gh = xin @ whh^T + bhh   (xin = prev state bf16)
// One wave: 16 rows x 16 cols, SIX accumulators (ir,iz,in,hr,hz,hn).
// 96 weight rows (48 KB) staged block-wide into LDS via async-to-LDS.
// grid = (HD/16, ceil(M/128)), block = 256.
// ---------------------------------------------------------------------------
__global__ __launch_bounds__(256)
void gru_fused_wmma(const bf16_t* __restrict__ hin, const bf16_t* __restrict__ xin,
                    const float* __restrict__ hprevF,
                    const bf16_t* __restrict__ wih, const bf16_t* __restrict__ whh,
                    const float* __restrict__ bih, const float* __restrict__ bhh,
                    float* __restrict__ xoutF, bf16_t* __restrict__ xoutB, int M) {
  __shared__ alignas(64) bf16_t smem[96 * HD];        // 96 rows x 256 bf16 = 48 KB
  const int wave = threadIdx.x >> 5;
  const int lane = threadIdx.x & 31;
  const int hf   = lane >> 4;
  const int l15  = lane & 15;
  const int j0 = blockIdx.x * 16;
  const int m0 = (blockIdx.y * 8 + wave) * 16;

  // ---- stage 96 weight rows into LDS (async, 12 x b128 per lane) ----
  // LDS rows 0..47  : wih[chunk*256 + j0 + r]  (chunk = r/16)
  // LDS rows 48..95 : whh[chunk*256 + j0 + r]
#pragma unroll
  for (int it = 0; it < 12; ++it) {
    int cch = (wave * 12 + it) * 32 + lane;           // chunk id 0..3071 (16B each)
    int r   = cch >> 5;                               // LDS row
    int o   = (cch & 31) * 8;                         // bf16 offset within row
    int rr  = (r < 48) ? r : r - 48;
    const bf16_t* wsrc = (r < 48) ? wih : whh;
    const bf16_t* g = wsrc + ((size_t)(rr >> 4) * HD + j0 + (rr & 15)) * HD + o;
    unsigned lds = (unsigned)(size_t)(&smem[(size_t)r * HD + o]);
    asm volatile("global_load_async_to_lds_b128 %0, %1, off"
                 :: "v"(lds), "v"(g) : "memory");
  }
  asm volatile("s_wait_asynccnt 0" ::: "memory");
  __syncthreads();

  if (m0 < M) {
    int rowA = m0 + l15; if (rowA >= M) rowA = M - 1;
    const bf16_t* pah = hin + (size_t)rowA * HD + hf * 16;
    const bf16_t* pax = xin + (size_t)rowA * HD + hf * 16;

    v8f c[6];
#pragma unroll
    for (int q = 0; q < 6; ++q)
#pragma unroll
      for (int r = 0; r < 8; ++r) c[q][r] = 0.f;

    for (int k0 = 0; k0 < HD; k0 += 32) {
      v16bf ah = load_afrag(pah, k0);
      v16bf ax = load_afrag(pax, k0);
#pragma unroll
      for (int q = 0; q < 3; ++q) {
        v16bf b = *(const v16bf*)(&smem[(size_t)(q * 16 + l15) * HD + k0 + hf * 16]);
        c[q] = __builtin_amdgcn_wmma_f32_16x16x32_bf16(false, ah, false, b,
                                                       (short)0, c[q], false, false);
      }
#pragma unroll
      for (int q = 0; q < 3; ++q) {
        v16bf b = *(const v16bf*)(&smem[(size_t)(48 + q * 16 + l15) * HD + k0 + hf * 16]);
        c[3 + q] = __builtin_amdgcn_wmma_f32_16x16x32_bf16(false, ax, false, b,
                                                           (short)0, c[3 + q], false, false);
      }
    }

    const int col = j0 + l15;
    const float b_ir = bih[col], b_iz = bih[HD + col], b_in = bih[2 * HD + col];
    const float b_hr = bhh[col], b_hz = bhh[HD + col], b_hn = bhh[2 * HD + col];
#pragma unroll
    for (int r = 0; r < 8; ++r) {
      int row = m0 + hf * 8 + r;
      if (row < M) {
        float ir = c[0][r] + b_ir, iz = c[1][r] + b_iz, inn = c[2][r] + b_in;
        float hr = c[3][r] + b_hr, hz = c[4][r] + b_hz, hn  = c[5][r] + b_hn;
        float rg = 1.f / (1.f + expf(-(ir + hr)));
        float zg = 1.f / (1.f + expf(-(iz + hz)));
        float ng = tanhf(inn + rg * hn);
        float v  = (1.f - zg) * ng + zg * hprevF[(size_t)row * HD + col];
        v = v > 0.f ? v : 0.f;
        xoutF[(size_t)row * HD + col] = v;
        xoutB[(size_t)row * HD + col] = (bf16_t)v;
      }
    }
  }
}

// ---------------------------------------------------------------------------
// Fused GATEConv edge attention:
//   a[e] = lrelu( sum_n lrelu( (W1 [x[src];eattr])[n] ) * att_l[n] + xr[dst] )
// ---------------------------------------------------------------------------
__global__ __launch_bounds__(256)
void gate_edge_kernel(const bf16_t* __restrict__ xbf,   // [N,256] bf16
                      const float* __restrict__ eattr,  // [E]
                      const bf16_t* __restrict__ Wg,    // [256,288] bf16 padded
                      const float* __restrict__ attl,   // [256]
                      const float* __restrict__ xr,     // [N]  x @ g_att_r
                      const int* __restrict__ srcI, const int* __restrict__ dstI,
                      float* __restrict__ a_out, int E) {
  const int wave = threadIdx.x >> 5;
  const int lane = threadIdx.x & 31;
  const int hf   = lane >> 4;
  const int l15  = lane & 15;
  const int e0 = (blockIdx.x * 8 + wave) * 16;
  if (e0 >= E) return;

  int e = e0 + l15; if (e >= E) e = E - 1;
  const int s = srcI[e];
  const bf16_t* pa = xbf + (size_t)s * HD + hf * 16;

  v16bf afr[8];
#pragma unroll
  for (int kt = 0; kt < 8; ++kt) afr[kt] = load_afrag(pa, kt * 32);
  v16bf alast;
#pragma unroll
  for (int i = 0; i < 16; ++i) alast[i] = (bf16_t)0.0f;
  if (hf == 0) alast[0] = (bf16_t)eattr[e];   // K=256 carries edge_attr

  float acc[8];
#pragma unroll
  for (int r = 0; r < 8; ++r) acc[r] = 0.f;

  for (int n0 = 0; n0 < HD; n0 += 16) {
    const bf16_t* pb = Wg + (size_t)(n0 + l15) * 288 + hf * 16;
    v8f c = {0.f,0.f,0.f,0.f,0.f,0.f,0.f,0.f};
#pragma unroll
    for (int kt = 0; kt < 8; ++kt) {
      v16bf b = *(const v16bf*)(pb + kt * 32);
      c = __builtin_amdgcn_wmma_f32_16x16x32_bf16(false, afr[kt], false, b,
                                                  (short)0, c, false, false);
    }
    {
      v16bf b = *(const v16bf*)(pb + 256);
      c = __builtin_amdgcn_wmma_f32_16x16x32_bf16(false, alast, false, b,
                                                  (short)0, c, false, false);
    }
    const float attv = attl[n0 + l15];
#pragma unroll
    for (int r = 0; r < 8; ++r) {
      float g = c[r]; g = g >= 0.f ? g : SLOPE * g;  // lrelu(ge)
      acc[r] += g * attv;
    }
  }
#pragma unroll
  for (int r = 0; r < 8; ++r)
    for (int m = 8; m >= 1; m >>= 1)
      acc[r] += __shfl_xor(acc[r], m, 32);
  if (l15 == 0) {
#pragma unroll
    for (int r = 0; r < 8; ++r) {
      int ee = e0 + hf * 8 + r;
      if (ee < E) {
        float v = acc[r] + xr[dstI[ee]];
        a_out[ee] = v >= 0.f ? v : SLOPE * v;
      }
    }
  }
}

// ---------------------------------------------------------------------------
// small elementwise / reduction / scatter kernels
// ---------------------------------------------------------------------------
__global__ void fill_u32_k(unsigned* p, unsigned v, int n) {
  int i = blockIdx.x * blockDim.x + threadIdx.x; if (i < n) p[i] = v;
}
__global__ void fill_f32_k(float* p, float v, long long n) {
  long long i = (long long)blockIdx.x * blockDim.x + threadIdx.x; if (i < n) p[i] = v;
}
__global__ void rowdot_k(const float* __restrict__ X, const float* __restrict__ v,
                         float* __restrict__ out, int R, int C) {
  int row = blockIdx.x * 8 + (threadIdx.x >> 5);
  int lane = threadIdx.x & 31;
  if (row >= R) return;
  float s = 0.f;
  for (int j = lane; j < C; j += 32) s += X[(size_t)row * C + j] * v[j];
  for (int m = 16; m >= 1; m >>= 1) s += __shfl_xor(s, m, 32);
  if (lane == 0) out[row] = s;
}
__global__ void edge_a_k(const float* asrc, const float* adst,
                         const int* srcI, const int* dstI, float* a, int E) {
  int e = blockIdx.x * blockDim.x + threadIdx.x; if (e >= E) return;
  float v = asrc[srcI[e]] + adst[dstI[e]];
  a[e] = v >= 0.f ? v : SLOPE * v;
}
__global__ void mol_a_k(const float* asrcN, const float* adstG,
                        const int* batch, float* a, int N) {
  int i = blockIdx.x * blockDim.x + threadIdx.x; if (i >= N) return;
  float v = asrcN[i] + adstG[batch[i]];
  a[i] = v >= 0.f ? v : SLOPE * v;
}
__global__ void seg_amax_k(const float* a, const int* idx, unsigned* menc, int n) {
  int i = blockIdx.x * blockDim.x + threadIdx.x; if (i >= n) return;
  atomicMax(&menc[idx[i]], encf(a[i]));
}
__global__ void seg_exp_k(float* a, const int* idx, const unsigned* menc,
                          float* s, int n) {
  int i = blockIdx.x * blockDim.x + threadIdx.x; if (i >= n) return;
  float e = expf(a[i] - decf(menc[idx[i]]));
  a[i] = e;
  atomicAdd(&s[idx[i]], e);
}
__global__ void init_rows_bias_k(float* out, const float* bias, long long rows, int C) {
  long long t = (long long)blockIdx.x * blockDim.x + threadIdx.x;
  if (t >= rows * C) return;
  out[t] = bias[t % C];
}
__global__ void seg_scatter_k(const float* __restrict__ X, const int* __restrict__ gidx,
                              const int* __restrict__ sidx, const float* __restrict__ ev,
                              const float* __restrict__ ss, float* __restrict__ acc,
                              long long n, int C) {
  long long t = (long long)blockIdx.x * blockDim.x + threadIdx.x;
  if (t >= n * C) return;
  long long e = t / C; int j = (int)(t % C);
  int sg = sidx[e];
  float alpha = ev[e] / ss[sg];
  long long srow = gidx ? (long long)gidx[e] : e;
  atomicAdd(&acc[(long long)sg * C + j], X[srow * C + j] * alpha);
}
__global__ void pool_sum_k(const float* x, const int* batch, float* pool,
                           long long n, int C) {
  long long t = (long long)blockIdx.x * blockDim.x + threadIdx.x;
  if (t >= n * C) return;
  long long r = t / C; int j = (int)(t % C);
  atomicAdd(&pool[(long long)batch[r] * C + j], x[t]);
}
__global__ void relu_cvt_k(const float* in, float* outf, bf16_t* outb, long long n) {
  long long t = (long long)blockIdx.x * blockDim.x + threadIdx.x;
  if (t >= n) return;
  float v = in[t]; v = v > 0.f ? v : 0.f;
  outf[t] = v; outb[t] = (bf16_t)v;
}
__global__ void elu_cvt_k(const float* in, bf16_t* outb, long long n) {
  long long t = (long long)blockIdx.x * blockDim.x + threadIdx.x;
  if (t >= n) return;
  float v = in[t];
  v = v > 0.f ? v : (expf(v) - 1.0f);
  outb[t] = (bf16_t)v;
}

// ---------------------------------------------------------------------------
static inline unsigned cdivu(long long a, long long b) { return (unsigned)((a + b - 1) / b); }

extern "C" void kernel_launch(void* const* d_in, const int* in_sizes, int n_in,
                              void* d_out, int out_size, void* d_ws, size_t ws_size,
                              hipStream_t stream) {
  (void)n_in; (void)out_size; (void)ws_size;
  const float* x_in     = (const float*)d_in[0];
  const float* eattr    = (const float*)d_in[1];
  const float* lin1_W   = (const float*)d_in[2];
  const float* lin1_b   = (const float*)d_in[3];
  const float* glin1_W  = (const float*)d_in[4];
  const float* glin2_W  = (const float*)d_in[5];
  const float* g_att_l  = (const float*)d_in[6];
  const float* g_att_r  = (const float*)d_in[7];
  const float* g_bias   = (const float*)d_in[8];
  const float* gru0_wih = (const float*)d_in[9];
  const float* gru0_whh = (const float*)d_in[10];
  const float* gru0_bih = (const float*)d_in[11];
  const float* gru0_bhh = (const float*)d_in[12];
  const float* atom_W   = (const float*)d_in[13];
  const float* a_att_s  = (const float*)d_in[14];
  const float* a_att_d  = (const float*)d_in[15];
  const float* a_bias   = (const float*)d_in[16];
  const float* agru_wih = (const float*)d_in[17];
  const float* agru_whh = (const float*)d_in[18];
  const float* agru_bih = (const float*)d_in[19];
  const float* agru_bhh = (const float*)d_in[20];
  const float* mol_Wp   = (const float*)d_in[21];
  const float* m_att_s  = (const float*)d_in[22];
  const float* m_att_d  = (const float*)d_in[23];
  const float* mol_b    = (const float*)d_in[24];
  const float* mgru_wih = (const float*)d_in[25];
  const float* mgru_whh = (const float*)d_in[26];
  const float* mgru_bih = (const float*)d_in[27];
  const float* mgru_bhh = (const float*)d_in[28];
  const float* lin2_W   = (const float*)d_in[29];
  const float* lin2_b   = (const float*)d_in[30];
  const float* fc1_W    = (const float*)d_in[31];
  const float* fc1_b    = (const float*)d_in[32];
  const int*   eidx     = (const int*)d_in[33];
  const int*   batch    = (const int*)d_in[34];

  const int Nn = in_sizes[34];          // nodes
  const int Ee = in_sizes[1];           // edges
  const int* srcI = eidx;
  const int* dstI = eidx + Ee;

  // ---------------- workspace bump allocator ----------------
  char* base = (char*)d_ws; size_t off = 0;
  auto alloc = [&](size_t bytes) -> void* {
    void* p = base + off; off = (off + bytes + 255) & ~(size_t)255; return p;
  };
  // bf16 weight buffers (K padded to multiple of 32)
  bf16_t* wb_lin1 = (bf16_t*)alloc((size_t)HD * 320 * 2);
  bf16_t* wb_g1   = (bf16_t*)alloc((size_t)HD * 288 * 2);
  bf16_t* wb_g2   = (bf16_t*)alloc((size_t)HD * HD * 2);
  bf16_t* wb_g0ih = (bf16_t*)alloc((size_t)3 * HD * HD * 2);
  bf16_t* wb_g0hh = (bf16_t*)alloc((size_t)3 * HD * HD * 2);
  bf16_t* wb_aW   = (bf16_t*)alloc((size_t)3 * HD * HD * 2);
  bf16_t* wb_aih  = (bf16_t*)alloc((size_t)3 * 3 * HD * HD * 2);
  bf16_t* wb_ahh  = (bf16_t*)alloc((size_t)3 * 3 * HD * HD * 2);
  bf16_t* wb_mW   = (bf16_t*)alloc((size_t)HD * HD * 2);
  bf16_t* wb_mih  = (bf16_t*)alloc((size_t)3 * HD * HD * 2);
  bf16_t* wb_mhh  = (bf16_t*)alloc((size_t)3 * HD * HD * 2);
  bf16_t* wb_l2   = (bf16_t*)alloc((size_t)512 * HD * 2);
  bf16_t* wb_fc1  = (bf16_t*)alloc((size_t)768 * 512 * 2);
  // activations (state is double-buffered for the fused GRU)
  bf16_t* xbf_in = (bf16_t*)alloc((size_t)Nn * 320 * 2);
  float*  xF     = (float*) alloc((size_t)Nn * HD * 4);
  bf16_t* xB     = (bf16_t*)alloc((size_t)Nn * HD * 2);
  float*  xF2    = (float*) alloc((size_t)Nn * HD * 4);
  bf16_t* xB2    = (bf16_t*)alloc((size_t)Nn * HD * 2);
  float*  hF     = (float*) alloc((size_t)Nn * HD * 4);
  bf16_t* hB     = (bf16_t*)alloc((size_t)Nn * HD * 2);
  float*  xw     = (float*) alloc((size_t)Nn * HD * 4);
  float*  aE     = (float*) alloc((size_t)Ee * 4);
  unsigned* segM = (unsigned*)alloc((size_t)Nn * 4);
  float*  segS   = (float*) alloc((size_t)Nn * 4);
  float*  xr     = (float*) alloc((size_t)Nn * 4);
  float*  asrc   = (float*) alloc((size_t)Nn * 4);
  float*  adst   = (float*) alloc((size_t)Nn * 4);
  float*  outF   = (float*) alloc((size_t)GN * HD * 4);
  bf16_t* outB   = (bf16_t*)alloc((size_t)GN * HD * 2);
  float*  outF2  = (float*) alloc((size_t)GN * HD * 4);
  bf16_t* outB2  = (bf16_t*)alloc((size_t)GN * HD * 2);
  float*  tmpg   = (float*) alloc((size_t)GN * HD * 4);
  float*  adstg  = (float*) alloc((size_t)GN * 4);
  float*  hg     = (float*) alloc((size_t)GN * HD * 4);
  bf16_t* hgB    = (bf16_t*)alloc((size_t)GN * HD * 2);
  float*  out2   = (float*) alloc((size_t)GN * 512 * 4);
  bf16_t* out2B  = (bf16_t*)alloc((size_t)GN * 512 * 2);

  // ---------------- launch helpers ----------------
  auto cvt = [&](const float* s, bf16_t* d, long long R, int C, int Cp) {
    long long total = R * (long long)Cp;
    cvt_pad_bf16<<<cdivu(total, 256), 256, 0, stream>>>(s, d, R, C, Cp);
  };
  auto gemm = [&](const bf16_t* A, const bf16_t* W, const float* bias, float* C,
                  bf16_t* Cb, int M, int Ncol, int K, int act) {
    dim3 gr(Ncol / 64, cdivu(M, 128));
    gemm4_bf16_wmma<<<gr, 256, 0, stream>>>(A, W, bias, C, Cb, M, Ncol, K, act);
  };
  auto gru = [&](const bf16_t* hin, const bf16_t* xin, const float* hprevF,
                 const bf16_t* wih, const bf16_t* whh, const float* bih,
                 const float* bhh, float* oF, bf16_t* oB, int M) {
    dim3 gr(HD / 16, cdivu(M, 128));
    gru_fused_wmma<<<gr, 256, 0, stream>>>(hin, xin, hprevF, wih, whh, bih, bhh, oF, oB, M);
  };
  auto rowdot = [&](const float* X, const float* v, float* o, int R, int C) {
    rowdot_k<<<cdivu(R, 8), 256, 0, stream>>>(X, v, o, R, C);
  };
  auto segmax_reset = [&](int nseg) {
    fill_u32_k<<<cdivu(nseg, 256), 256, 0, stream>>>(segM, ENC_NEG_INF, nseg);
    fill_f32_k<<<cdivu(nseg, 256), 256, 0, stream>>>(segS, 0.f, nseg);
  };
  auto softmax = [&](float* a, const int* idx, int n) {
    seg_amax_k<<<cdivu(n, 256), 256, 0, stream>>>(a, idx, segM, n);
    seg_exp_k<<<cdivu(n, 256), 256, 0, stream>>>(a, idx, segM, segS, n);
  };

  // ---------------- weight conversion (fp32 -> padded bf16) ----------------
  cvt(lin1_W,   wb_lin1, HD, 300, 320);
  cvt(glin1_W,  wb_g1,   HD, 257, 288);
  cvt(glin2_W,  wb_g2,   HD, HD, HD);
  cvt(gru0_wih, wb_g0ih, 3 * HD, HD, HD);
  cvt(gru0_whh, wb_g0hh, 3 * HD, HD, HD);
  cvt(atom_W,   wb_aW,   3 * HD, HD, HD);
  cvt(agru_wih, wb_aih,  3 * 3 * HD, HD, HD);
  cvt(agru_whh, wb_ahh,  3 * 3 * HD, HD, HD);
  cvt(mol_Wp,   wb_mW,   HD, HD, HD);
  cvt(mgru_wih, wb_mih,  3 * HD, HD, HD);
  cvt(mgru_whh, wb_mhh,  3 * HD, HD, HD);
  cvt(lin2_W,   wb_l2,   512, HD, HD);
  cvt(fc1_W,    wb_fc1,  768, 512, 512);

  // current / alternate state pointers (node level)
  float* curF = xF;  bf16_t* curB = xB;
  float* altF = xF2; bf16_t* altB = xB2;

  // ---------------- input projection: x = lrelu(x @ lin1^T + b) ----------------
  cvt(x_in, xbf_in, Nn, 300, 320);
  gemm(xbf_in, wb_lin1, lin1_b, curF, curB, Nn, HD, 320, /*lrelu*/1);

  // ---------------- GATEConv + GRU0 ----------------
  rowdot(curF, g_att_r, xr, Nn, HD);
  gate_edge_kernel<<<cdivu(Ee, 128), 256, 0, stream>>>(curB, eattr, wb_g1, g_att_l,
                                                       xr, srcI, dstI, aE, Ee);
  segmax_reset(Nn);
  softmax(aE, dstI, Ee);
  gemm(curB, wb_g2, nullptr, xw, nullptr, Nn, HD, HD, 0);    // xw2 = x @ g_lin2^T
  init_rows_bias_k<<<cdivu((long long)Nn * HD, 256), 256, 0, stream>>>(hF, g_bias, Nn, HD);
  seg_scatter_k<<<cdivu((long long)Ee * HD, 256), 256, 0, stream>>>(
      xw, srcI, dstI, aE, segS, hF, Ee, HD);
  elu_cvt_k<<<cdivu((long long)Nn * HD, 256), 256, 0, stream>>>(hF, hB, (long long)Nn * HD);
  gru(hB, curB, curF, wb_g0ih, wb_g0hh, gru0_bih, gru0_bhh, altF, altB, Nn);
  { float* tf = curF; curF = altF; altF = tf; bf16_t* tb = curB; curB = altB; altB = tb; }

  // ---------------- 3 atom GATConv + GRU layers ----------------
  for (int l = 0; l < 3; ++l) {
    gemm(curB, wb_aW + (size_t)l * HD * HD, nullptr, xw, nullptr, Nn, HD, HD, 0);
    rowdot(xw, a_att_s + l * HD, asrc, Nn, HD);
    rowdot(xw, a_att_d + l * HD, adst, Nn, HD);
    edge_a_k<<<cdivu(Ee, 256), 256, 0, stream>>>(asrc, adst, srcI, dstI, aE, Ee);
    segmax_reset(Nn);
    softmax(aE, dstI, Ee);
    init_rows_bias_k<<<cdivu((long long)Nn * HD, 256), 256, 0, stream>>>(hF, a_bias + l * HD, Nn, HD);
    seg_scatter_k<<<cdivu((long long)Ee * HD, 256), 256, 0, stream>>>(
        xw, srcI, dstI, aE, segS, hF, Ee, HD);
    elu_cvt_k<<<cdivu((long long)Nn * HD, 256), 256, 0, stream>>>(hF, hB, (long long)Nn * HD);
    gru(hB, curB, curF, wb_aih + (size_t)l * 3 * HD * HD, wb_ahh + (size_t)l * 3 * HD * HD,
        agru_bih + l * 768, agru_bhh + l * 768, altF, altB, Nn);
    { float* tf = curF; curF = altF; altF = tf; bf16_t* tb = curB; curB = altB; altB = tb; }
  }

  // ---------------- attentive global pooling (16 timesteps) ----------------
  float* ocF = outF;  bf16_t* ocB = outB;
  float* oaF = outF2; bf16_t* oaB = outB2;
  fill_f32_k<<<cdivu((long long)GN * HD, 256), 256, 0, stream>>>(ocF, 0.f, (long long)GN * HD);
  pool_sum_k<<<cdivu((long long)Nn * HD, 256), 256, 0, stream>>>(curF, batch, ocF, Nn, HD);
  relu_cvt_k<<<cdivu((long long)GN * HD, 256), 256, 0, stream>>>(ocF, ocF, ocB, (long long)GN * HD);
  gemm(curB, wb_mW, nullptr, xw, nullptr, Nn, HD, HD, 0);    // xw_mol (reused all steps)
  rowdot(xw, m_att_s, asrc, Nn, HD);
  for (int t = 0; t < 16; ++t) {
    gemm(ocB, wb_mW, nullptr, tmpg, nullptr, GN, HD, HD, 0);
    rowdot(tmpg, m_att_d, adstg, GN, HD);
    mol_a_k<<<cdivu(Nn, 256), 256, 0, stream>>>(asrc, adstg, batch, aE, Nn);
    segmax_reset(GN);
    softmax(aE, batch, Nn);
    init_rows_bias_k<<<cdivu((long long)GN * HD, 256), 256, 0, stream>>>(hg, mol_b, GN, HD);
    seg_scatter_k<<<cdivu((long long)Nn * HD, 256), 256, 0, stream>>>(
        xw, nullptr, batch, aE, segS, hg, Nn, HD);
    elu_cvt_k<<<cdivu((long long)GN * HD, 256), 256, 0, stream>>>(hg, hgB, (long long)GN * HD);
    gru(hgB, ocB, ocF, wb_mih, wb_mhh, mgru_bih, mgru_bhh, oaF, oaB, GN);
    { float* tf = ocF; ocF = oaF; oaF = tf; bf16_t* tb = ocB; ocB = oaB; oaB = tb; }
  }

  // ---------------- head: relu(out @ lin2^T + b) @ fc1^T + b ----------------
  gemm(ocB, wb_l2, lin2_b, out2, out2B, GN, 512, HD, /*relu*/2);
  gemm(out2B, wb_fc1, fc1_b, (float*)d_out, nullptr, GN, 768, 512, 0);
}
